// MoE_57492432224434
// MI455X (gfx1250) — compile-verified, matrix operand order checked by reading, not run on hardware
//
#include <hip/hip_runtime.h>

// ---------------------------------------------------------------------------
// MoE (top-2 of 8 experts, SwiGLU) for MI455X / gfx1250.
// Sparse-routing formulation of the dense reference (combine weights are zero
// for unselected experts, so only T*K rows of expert work are needed).
// Expert GEMMs: v_wmma_f32_16x16x32_bf16 (bf16 operands, fp32 accumulate).
// Global->LDS staging: GLOBAL_LOAD_ASYNC_TO_LDS_B128 with a 3-stage rotating
// buffer and partial s_wait_asynccnt waits (loads complete in order), one
// workgroup barrier per K-step.
// ---------------------------------------------------------------------------

namespace {

constexpr int kD  = 1024;   // model dim
constexpr int kF  = 2816;   // expert hidden dim
constexpr int kE  = 8;      // experts
constexpr int kT  = 4096;   // tokens (B*L = 2*2048)
constexpr int kTK = 8192;   // tokens * top-k

typedef __bf16 bf16;
typedef __attribute__((ext_vector_type(8)))  __bf16 v8bf;
typedef __attribute__((ext_vector_type(16))) __bf16 v16bf;
typedef __attribute__((ext_vector_type(8)))  float  v8f;

// LDS row stride in halves: 80 bytes -> 16B aligned and bank-conflict-free
// for 16-lane row-parallel b128 reads (20*r mod 64 hits 16 distinct banks).
constexpr int kStride = 40;

__device__ inline uint32_t lds_off(const void* p) {
  // addrspace(3) pointer value == LDS byte offset; flat cast keeps it in
  // the low 32 bits (aperture lives in the high word).
  return (uint32_t)(uintptr_t)p;
}

// CDNA5 async global->LDS copy, 16B per lane. Tracked with ASYNCcnt.
__device__ inline void async_load_b128(uint32_t lds_addr, const void* gaddr) {
  asm volatile("global_load_async_to_lds_b128 %0, %1, off"
               :: "v"(lds_addr), "v"((uint64_t)(uintptr_t)gaddr)
               : "memory");
}

__device__ inline void wait_async_le0() {
  asm volatile("s_wait_asynccnt 0x0" ::: "memory");
}
// Leave the most recent 5 async loads (next stage) in flight; the oldest 5
// (current stage) are guaranteed complete because async loads retire in order.
__device__ inline void wait_async_le5() {
  asm volatile("s_wait_asynccnt 0x5" ::: "memory");
}

__device__ inline v16bf lds_frag(const bf16* base, int row, int kbase) {
  const bf16* p = base + row * kStride + kbase;
  v8bf lo = *reinterpret_cast<const v8bf*>(p);        // K = kbase .. kbase+7
  v8bf hi = *reinterpret_cast<const v8bf*>(p + 16);   // K = kbase+16 .. +23
  return __builtin_shufflevector(lo, hi, 0, 1, 2, 3, 4, 5, 6, 7,
                                 8, 9, 10, 11, 12, 13, 14, 15);
}

// ---------------------------------------------------------------------------
// fp32 -> bf16 cast, 8 elements / thread
// ---------------------------------------------------------------------------
__global__ void cast_bf16_kernel(const float* __restrict__ s,
                                 bf16* __restrict__ d, int n8) {
  int i = blockIdx.x * blockDim.x + threadIdx.x;
  if (i >= n8) return;
  const float4 a = *reinterpret_cast<const float4*>(s + (size_t)i * 8);
  const float4 b = *reinterpret_cast<const float4*>(s + (size_t)i * 8 + 4);
  v8bf o;
  o[0] = (bf16)a.x; o[1] = (bf16)a.y; o[2] = (bf16)a.z; o[3] = (bf16)a.w;
  o[4] = (bf16)b.x; o[5] = (bf16)b.y; o[6] = (bf16)b.z; o[7] = (bf16)b.w;
  *reinterpret_cast<v8bf*>(d + (size_t)i * 8) = o;
}

// ---------------------------------------------------------------------------
// init / scan / scatter for routing metadata
// ---------------------------------------------------------------------------
__global__ void init_cnt_kernel(int* __restrict__ cnt) {
  if (threadIdx.x < kE) cnt[threadIdx.x] = 0;
}

__global__ void exscan_kernel(const int* __restrict__ cnt,
                              int* __restrict__ off,
                              int* __restrict__ cursor) {
  if (threadIdx.x == 0) {
    int s = 0;
    for (int e = 0; e < kE; ++e) { off[e] = s; s += cnt[e]; }
    off[kE] = s;
  }
  if (threadIdx.x < kE) cursor[threadIdx.x] = 0;
}

// ---------------------------------------------------------------------------
// Router: one wave per token. logits = x @ gw^T, top-2, softmax(gates).
// ---------------------------------------------------------------------------
__global__ __launch_bounds__(256)
void router_kernel(const float* __restrict__ x, const float* __restrict__ gw,
                   int* __restrict__ tokInfo, float* __restrict__ tokGate,
                   int* __restrict__ cnt) {
  const int wave = threadIdx.x >> 5;
  const int lane = threadIdx.x & 31;
  const int t = blockIdx.x * 8 + wave;
  const float* xr = x + (size_t)t * kD;

  float acc[kE];
#pragma unroll
  for (int e = 0; e < kE; ++e) acc[e] = 0.f;

#pragma unroll
  for (int i = 0; i < 8; ++i) {
    const int d = lane * 4 + i * 128;
    const float4 xv = *reinterpret_cast<const float4*>(xr + d);
#pragma unroll
    for (int e = 0; e < kE; ++e) {
      const float4 gv = *reinterpret_cast<const float4*>(gw + e * kD + d);
      acc[e] += xv.x * gv.x + xv.y * gv.y + xv.z * gv.z + xv.w * gv.w;
    }
  }
#pragma unroll
  for (int m = 16; m >= 1; m >>= 1)
#pragma unroll
    for (int e = 0; e < kE; ++e) acc[e] += __shfl_xor(acc[e], m, 32);

  if (lane == 0) {
    int e0 = 0; float v0 = acc[0];
#pragma unroll
    for (int e = 1; e < kE; ++e)
      if (acc[e] > v0) { v0 = acc[e]; e0 = e; }
    int e1 = -1; float v1 = -3.0e38f;
#pragma unroll
    for (int e = 0; e < kE; ++e)
      if (e != e0 && acc[e] > v1) { v1 = acc[e]; e1 = e; }

    const float ex  = __expf(v1 - v0);
    const float inv = 1.f / (1.f + ex);
    tokInfo[t]        = e0 | (e1 << 4);
    tokGate[2 * t]     = inv;        // gate for top-1
    tokGate[2 * t + 1] = ex * inv;   // gate for top-2
    atomicAdd(&cnt[e0], 1);
    atomicAdd(&cnt[e1], 1);
  }
}

__global__ void scatter_kernel(const int* __restrict__ tokInfo,
                               const float* __restrict__ tokGate,
                               const int* __restrict__ off,
                               int* __restrict__ cursor,
                               int* __restrict__ rowTok,
                               float* __restrict__ rowGate) {
  const int t = blockIdx.x * blockDim.x + threadIdx.x;
  if (t >= kT) return;
  const int info = tokInfo[t];
#pragma unroll
  for (int k = 0; k < 2; ++k) {
    const int e = (info >> (4 * k)) & 0xF;
    const int pos = atomicAdd(&cursor[e], 1);
    const int idx = off[e] + pos;
    rowTok[idx]  = t * 2 + k;          // (token, which-k) -> ybuf slot
    rowGate[idx] = tokGate[t * 2 + k];
  }
}

// ---------------------------------------------------------------------------
// GEMM1 (fused SwiGLU): H[row, f] = silu(x . w1[e,f]) * (x . w2[e,f])
// Block tile: M=64 rows x N=128 features, K-step 32 over D (32 steps).
// 8 waves: 4 (M) x 2 (N); each wave owns 16x64 of C for both w1 and w2.
// 3-stage rotating async LDS staging, one barrier per K-step.
// ---------------------------------------------------------------------------
__global__ __launch_bounds__(256)
void gemm1_swiglu_kernel(const bf16* __restrict__ xb,
                         const bf16* __restrict__ w1b,
                         const bf16* __restrict__ w2b,
                         const int* __restrict__ cnt,
                         const int* __restrict__ off,
                         const int* __restrict__ rowTok,
                         bf16* __restrict__ hbuf) {
  const int e  = blockIdx.y;
  const int c  = cnt[e];
  const int m0 = blockIdx.x * 64;
  if (m0 >= c) return;
  const int base = off[e];
  const int n0 = blockIdx.z * 128;
  constexpr int kNI = kD / 32;            // 32 K-steps
  constexpr uint32_t kABuf = 64  * kStride * sizeof(bf16);
  constexpr uint32_t kBBuf = 128 * kStride * sizeof(bf16);

  __shared__ __align__(16) bf16 shA [3][64  * kStride];
  __shared__ __align__(16) bf16 shB1[3][128 * kStride];
  __shared__ __align__(16) bf16 shB2[3][128 * kStride];

  const int tid = threadIdx.x;

  // A loader: row = tid>>2 (0..63), 16B chunk = tid&3 (gathered token row)
  const int aRowL = tid >> 2;
  int agrow = m0 + aRowL; if (agrow > c - 1) agrow = c - 1;
  const int atok = rowTok[base + agrow] >> 1;
  const int bc  = tid & 3;
  const bf16* aSrc = xb + (size_t)atok * kD + bc * 8;

  // B loaders: 128 rows x 4 chunks = 512 units over 256 threads (2 each)
  const int br0 = tid >> 2;           // rows 0..63
  const int br1 = br0 + 64;           // rows 64..127
  const bf16* b1s0 = w1b + ((size_t)e * kF + n0 + br0) * kD + bc * 8;
  const bf16* b1s1 = w1b + ((size_t)e * kF + n0 + br1) * kD + bc * 8;
  const bf16* b2s0 = w2b + ((size_t)e * kF + n0 + br0) * kD + bc * 8;
  const bf16* b2s1 = w2b + ((size_t)e * kF + n0 + br1) * kD + bc * 8;

  const uint32_t aDst0  = lds_off(&shA [0][aRowL * kStride + bc * 8]);
  const uint32_t b1d00  = lds_off(&shB1[0][br0   * kStride + bc * 8]);
  const uint32_t b1d10  = lds_off(&shB1[0][br1   * kStride + bc * 8]);
  const uint32_t b2d00  = lds_off(&shB2[0][br0   * kStride + bc * 8]);
  const uint32_t b2d10  = lds_off(&shB2[0][br1   * kStride + bc * 8]);

  const int wave = tid >> 5, lane = tid & 31;
  const int wM = wave >> 1, wN = wave & 1;
  const int aRow = wM * 16 + (lane & 15);
  const int kb = (lane >> 4) * 8;

  v8f acc1[4] = {}, acc2[4] = {};

  auto issue = [&](int bi, int k0) {                 // 5 async loads / thread
    async_load_b128(aDst0 + bi * kABuf, aSrc  + k0);
    async_load_b128(b1d00 + bi * kBBuf, b1s0 + k0);
    async_load_b128(b1d10 + bi * kBBuf, b1s1 + k0);
    async_load_b128(b2d00 + bi * kBBuf, b2s0 + k0);
    async_load_b128(b2d10 + bi * kBBuf, b2s1 + k0);
  };

  issue(0, 0);
  issue(1, 32);
  int bi = 0;
  for (int i = 0; i < kNI; ++i) {
    if (i + 1 < kNI) wait_async_le5(); else wait_async_le0();
    __syncthreads();  // publish buffer bi; certify buffer (bi+2)%3 consumed
    if (i + 2 < kNI) issue(bi == 0 ? 2 : bi - 1, i * 32 + 64);

    const bf16* aT  = &shA [bi][0];
    const bf16* b1T = &shB1[bi][0];
    const bf16* b2T = &shB2[bi][0];
    const v16bf a = lds_frag(aT, aRow, kb);
#pragma unroll
    for (int j = 0; j < 4; ++j) {
      const int bRow = wN * 64 + j * 16 + (lane & 15);
      const v16bf b1 = lds_frag(b1T, bRow, kb);
      acc1[j] = __builtin_amdgcn_wmma_f32_16x16x32_bf16(
          false, a, false, b1, (short)0, acc1[j], false, false);
      const v16bf b2 = lds_frag(b2T, bRow, kb);
      acc2[j] = __builtin_amdgcn_wmma_f32_16x16x32_bf16(
          false, a, false, b2, (short)0, acc2[j], false, false);
    }
    bi = (bi == 2) ? 0 : bi + 1;
  }

  // Epilogue: h = silu(c1) * c2, store bf16 into compacted H buffer.
  const int n = lane & 15;
#pragma unroll
  for (int r = 0; r < 8; ++r) {
    const int mloc = wM * 16 + ((lane < 16) ? r : r + 8);
    const int grow = m0 + mloc;
    if (grow < c) {
#pragma unroll
      for (int j = 0; j < 4; ++j) {
        const float c1 = acc1[j][r];
        const float c2 = acc2[j][r];
        const float sig = __builtin_amdgcn_rcpf(1.f + __expf(-c1));
        const float h = c1 * sig * c2;
        const int fcol = n0 + wN * 64 + j * 16 + n;
        hbuf[(size_t)(base + grow) * kF + fcol] = (bf16)h;
      }
    }
  }
}

// ---------------------------------------------------------------------------
// GEMM2: ybuf[slot, d] = gate * sum_f H[row, f] * w3[e, d, f]
// Block tile: M=64 x N=256 over D, K = F = 2816 (88 steps of 32).
// 8 waves: 4 (M) x 2 (N); wave tile 16x128 (acc[8]).
// ---------------------------------------------------------------------------
__global__ __launch_bounds__(256)
void gemm2_down_kernel(const bf16* __restrict__ hbuf,
                       const bf16* __restrict__ w3b,
                       const int* __restrict__ cnt,
                       const int* __restrict__ off,
                       const int* __restrict__ rowTok,
                       const float* __restrict__ rowGate,
                       float* __restrict__ ybuf) {
  const int e  = blockIdx.y;
  const int c  = cnt[e];
  const int m0 = blockIdx.x * 64;
  if (m0 >= c) return;
  const int base = off[e];
  const int n0 = blockIdx.z * 256;
  constexpr int kNI = kF / 32;            // 88 K-steps
  constexpr uint32_t kABuf = 64  * kStride * sizeof(bf16);
  constexpr uint32_t kBBuf = 256 * kStride * sizeof(bf16);

  __shared__ __align__(16) bf16 shA[3][64  * kStride];
  __shared__ __align__(16) bf16 shB[3][256 * kStride];

  const int tid = threadIdx.x;

  const int aRowL = tid >> 2;
  int agrow = m0 + aRowL; if (agrow > c - 1) agrow = c - 1;
  const int bc  = tid & 3;
  const bf16* aSrc = hbuf + (size_t)(base + agrow) * kF + bc * 8;

  // B loaders: 256 rows x 4 chunks = 1024 units over 256 threads (4 each)
  const int br = tid >> 2;   // base row 0..63; units at +0,+64,+128,+192
  const bf16* bs0 = w3b + ((size_t)e * kD + n0 + br)       * kF + bc * 8;
  const bf16* bs1 = w3b + ((size_t)e * kD + n0 + br + 64)  * kF + bc * 8;
  const bf16* bs2 = w3b + ((size_t)e * kD + n0 + br + 128) * kF + bc * 8;
  const bf16* bs3 = w3b + ((size_t)e * kD + n0 + br + 192) * kF + bc * 8;

  const uint32_t aDst0 = lds_off(&shA[0][aRowL * kStride + bc * 8]);
  const uint32_t bd0   = lds_off(&shB[0][br          * kStride + bc * 8]);
  const uint32_t bd1   = lds_off(&shB[0][(br + 64)   * kStride + bc * 8]);
  const uint32_t bd2   = lds_off(&shB[0][(br + 128)  * kStride + bc * 8]);
  const uint32_t bd3   = lds_off(&shB[0][(br + 192)  * kStride + bc * 8]);

  const int wave = tid >> 5, lane = tid & 31;
  const int wM = wave >> 1, wN = wave & 1;
  const int aRow = wM * 16 + (lane & 15);
  const int kb = (lane >> 4) * 8;

  v8f acc[8] = {};

  auto issue = [&](int bi, int k0) {                 // 5 async loads / thread
    async_load_b128(aDst0 + bi * kABuf, aSrc + k0);
    async_load_b128(bd0 + bi * kBBuf, bs0 + k0);
    async_load_b128(bd1 + bi * kBBuf, bs1 + k0);
    async_load_b128(bd2 + bi * kBBuf, bs2 + k0);
    async_load_b128(bd3 + bi * kBBuf, bs3 + k0);
  };

  issue(0, 0);
  issue(1, 32);
  int bi = 0;
  for (int i = 0; i < kNI; ++i) {
    if (i + 1 < kNI) wait_async_le5(); else wait_async_le0();
    __syncthreads();
    if (i + 2 < kNI) issue(bi == 0 ? 2 : bi - 1, i * 32 + 64);

    const bf16* aT = &shA[bi][0];
    const bf16* bT = &shB[bi][0];
    const v16bf a = lds_frag(aT, aRow, kb);
#pragma unroll
    for (int j = 0; j < 8; ++j) {
      const int bRow = wN * 128 + j * 16 + (lane & 15);
      const v16bf b = lds_frag(bT, bRow, kb);
      acc[j] = __builtin_amdgcn_wmma_f32_16x16x32_bf16(
          false, a, false, b, (short)0, acc[j], false, false);
    }
    bi = (bi == 2) ? 0 : bi + 1;
  }

  const int n = lane & 15;
#pragma unroll
  for (int r = 0; r < 8; ++r) {
    const int mloc = wM * 16 + ((lane < 16) ? r : r + 8);
    const int grow = m0 + mloc;
    if (grow < c) {
      const int slot = rowTok[base + grow];     // t*2 + k  (unique per row)
      const float g = rowGate[base + grow];
#pragma unroll
      for (int j = 0; j < 8; ++j) {
        const int dcol = n0 + wN * 128 + j * 16 + n;
        ybuf[(size_t)slot * kD + dcol] = acc[j][r] * g;
      }
    }
  }
}

// ---------------------------------------------------------------------------
// Combine: y[t, d] = ybuf[2t, d] + ybuf[2t+1, d]   (deterministic, no atomics)
// ---------------------------------------------------------------------------
__global__ void combine_kernel(const float* __restrict__ ybuf,
                               float* __restrict__ out) {
  const size_t i = (size_t)(blockIdx.x * blockDim.x + threadIdx.x) * 4;
  const size_t t = i / kD;
  const size_t d = i % kD;
  const float4 a = *reinterpret_cast<const float4*>(ybuf + (2 * t) * kD + d);
  const float4 b = *reinterpret_cast<const float4*>(ybuf + (2 * t + 1) * kD + d);
  float4 r;
  r.x = a.x + b.x; r.y = a.y + b.y; r.z = a.z + b.z; r.w = a.w + b.w;
  *reinterpret_cast<float4*>(out + i) = r;
}

}  // namespace

// ---------------------------------------------------------------------------
// kernel_launch
// ---------------------------------------------------------------------------
extern "C" void kernel_launch(void* const* d_in, const int* in_sizes, int n_in,
                              void* d_out, int out_size, void* d_ws, size_t ws_size,
                              hipStream_t stream) {
  const float* xmat = (const float*)d_in[0];  // [T, D]
  const float* gw   = (const float*)d_in[1];  // [E, D]
  const float* w1   = (const float*)d_in[2];  // [E, F, D]
  const float* w2   = (const float*)d_in[3];  // [E, F, D]
  const float* w3   = (const float*)d_in[4];  // [E, D, F]
  float* out = (float*)d_out;

  // ---- carve workspace --------------------------------------------------
  char* p = (char*)d_ws;
  auto take = [&](size_t bytes) -> char* {
    char* r = p;
    p += (bytes + 255) & ~(size_t)255;
    return r;
  };
  int*   cnt     = (int*)take(kE * sizeof(int));
  int*   cursor  = (int*)take(kE * sizeof(int));
  int*   off     = (int*)take((kE + 1) * sizeof(int));
  int*   tokInfo = (int*)take((size_t)kT * sizeof(int));
  float* tokGate = (float*)take((size_t)kTK * sizeof(float));
  int*   rowTok  = (int*)take((size_t)kTK * sizeof(int));
  float* rowGate = (float*)take((size_t)kTK * sizeof(float));
  bf16*  xb      = (bf16*)take((size_t)kT * kD * sizeof(bf16));
  bf16*  w1b     = (bf16*)take((size_t)kE * kF * kD * sizeof(bf16));
  bf16*  w2b     = (bf16*)take((size_t)kE * kF * kD * sizeof(bf16));
  bf16*  w3b     = (bf16*)take((size_t)kE * kD * kF * sizeof(bf16));
  bf16*  hbuf    = (bf16*)take((size_t)kTK * kF * sizeof(bf16));
  float* ybuf    = (float*)take((size_t)kTK * kD * sizeof(float));
  (void)ws_size; (void)in_sizes; (void)n_in; (void)out_size;

  // ---- 1) fp32 -> bf16 casts -------------------------------------------
  {
    const int nx = kT * kD / 8;
    cast_bf16_kernel<<<nx / 256, 256, 0, stream>>>(xmat, xb, nx);
    const int nw = kE * kF * kD / 8;
    cast_bf16_kernel<<<nw / 256, 256, 0, stream>>>(w1, w1b, nw);
    cast_bf16_kernel<<<nw / 256, 256, 0, stream>>>(w2, w2b, nw);
    cast_bf16_kernel<<<nw / 256, 256, 0, stream>>>(w3, w3b, nw);
  }

  // ---- 2) routing -------------------------------------------------------
  init_cnt_kernel<<<1, 32, 0, stream>>>(cnt);
  router_kernel<<<kT / 8, 256, 0, stream>>>(xmat, gw, tokInfo, tokGate, cnt);
  exscan_kernel<<<1, 32, 0, stream>>>(cnt, off, cursor);
  scatter_kernel<<<kT / 256, 256, 0, stream>>>(tokInfo, tokGate, off, cursor,
                                               rowTok, rowGate);

  // ---- 3) expert GEMMs (WMMA bf16, 3-stage async LDS pipeline) ----------
  {
    dim3 g1(kT / 64, kE, kF / 128);   // (64, 8, 22); blocks past cnt[e] exit
    gemm1_swiglu_kernel<<<g1, 256, 0, stream>>>(xb, w1b, w2b, cnt, off, rowTok,
                                                hbuf);
    dim3 g2(kT / 64, kE, kD / 256);   // (64, 8, 4)
    gemm2_down_kernel<<<g2, 256, 0, stream>>>(hbuf, w3b, cnt, off, rowTok,
                                              rowGate, ybuf);
  }

  // ---- 4) combine the two expert contributions per token ---------------
  combine_kernel<<<(kT * kD / 4) / 256, 256, 0, stream>>>(ybuf, out);
}
